// QuantizedLinear_83949430767988
// MI455X (gfx1250) — matrix operand order repaired and checked by reading, not compiled, not run on hardware
//
#include <hip/hip_runtime.h>

typedef __attribute__((ext_vector_type(16))) __bf16 v16bf;
typedef __attribute__((ext_vector_type(8)))  __bf16 v8bf;
typedef __attribute__((ext_vector_type(8)))  float  v8f;
typedef __attribute__((ext_vector_type(4)))  float  v4f;
typedef __attribute__((ext_vector_type(4)))  int    v4i;

#define M_TOT 8192   // B*S = 4*2048
#define N_TOT 4096   // D_OUT
#define K_TOT 4096   // D_IN
#define BM 128
#define BN 128
#define BK 64
#define LDSS 72      // BK + 8 bf16 pad -> 36-dword row stride, conflict-free

__global__ __launch_bounds__(256, 1)
void qlin_wmma_bf16_kernel(const float* __restrict__ x,
                           const int*   __restrict__ wq,
                           const float* __restrict__ scale_p,
                           const float* __restrict__ zp_p,
                           const float* __restrict__ bias,
                           float*       __restrict__ out)
{
    __shared__ __bf16 ldsA[BM * LDSS];   // x tile, bf16, [m][k]
    __shared__ __bf16 ldsB[BN * LDSS];   // dequantized weight tile, bf16, [n][k]

    const int tid   = threadIdx.x;
    const int lane  = tid & 31;
    const int wv    = tid >> 5;      // 0..7
    const int waveM = wv >> 1;       // 0..3 -> 32-row strip
    const int waveN = wv & 1;        // 0..1 -> 64-col strip
    const int half  = lane >> 4;     // lane group
    const int l16   = lane & 15;

    const int nBase = blockIdx.x * BN;
    const int mBase = blockIdx.y * BM;

    const float s = scale_p[0];
    const float o = -zp_p[0] * s;    // w = q*s + o

    // ---- global->register staging: 2 threads per row, 32 elems each ----
    const int ldRow = tid >> 1;          // 0..127
    const int ldK   = (tid & 1) * 32;    // 0 or 32
    const float* xrow = x  + (size_t)(mBase + ldRow) * K_TOT + ldK;
    const int*   wrow = wq + (size_t)(nBase + ldRow) * K_TOT + ldK;

    v4f ra[8];
    v4i rb[8];

    v8f acc[2][4];
#pragma unroll
    for (int mf = 0; mf < 2; ++mf)
#pragma unroll
        for (int nf = 0; nf < 4; ++nf)
            acc[mf][nf] = (v8f)0.0f;

    const int nk = K_TOT / BK;  // 64

    auto loadTiles = [&](int kt) {
        const float* xp = xrow + kt * BK;
        const int*   wp = wrow + kt * BK;
#pragma unroll
        for (int i = 0; i < 8; ++i) {
            ra[i] = *(const v4f*)(xp + 4 * i);
            rb[i] = *(const v4i*)(wp + 4 * i);
        }
        if (kt + 1 < nk) {                       // warm L2 for tile kt+1
            __builtin_prefetch(xp + BK, 0, 1);
            __builtin_prefetch(wp + BK, 0, 1);
        }
    };

    auto stageTiles = [&]() {
        __bf16* pa = &ldsA[ldRow * LDSS + ldK];
        __bf16* pb = &ldsB[ldRow * LDSS + ldK];
#pragma unroll
        for (int i = 0; i < 4; ++i) {
            v4f f0 = ra[2 * i], f1 = ra[2 * i + 1];
            v8bf ta;
            ta[0] = (__bf16)f0[0]; ta[1] = (__bf16)f0[1];
            ta[2] = (__bf16)f0[2]; ta[3] = (__bf16)f0[3];
            ta[4] = (__bf16)f1[0]; ta[5] = (__bf16)f1[1];
            ta[6] = (__bf16)f1[2]; ta[7] = (__bf16)f1[3];
            *(v8bf*)(pa + 8 * i) = ta;

            v4i q0 = rb[2 * i], q1 = rb[2 * i + 1];
            v8bf tb;
            tb[0] = (__bf16)fmaf((float)q0[0], s, o);
            tb[1] = (__bf16)fmaf((float)q0[1], s, o);
            tb[2] = (__bf16)fmaf((float)q0[2], s, o);
            tb[3] = (__bf16)fmaf((float)q0[3], s, o);
            tb[4] = (__bf16)fmaf((float)q1[0], s, o);
            tb[5] = (__bf16)fmaf((float)q1[1], s, o);
            tb[6] = (__bf16)fmaf((float)q1[2], s, o);
            tb[7] = (__bf16)fmaf((float)q1[3], s, o);
            *(v8bf*)(pb + 8 * i) = tb;
        }
    };

    loadTiles(0);

    for (int kt = 0; kt < nk; ++kt) {
        __syncthreads();               // prior compute done, safe to overwrite LDS
        stageTiles();
        __syncthreads();               // LDS tile visible to all waves
        if (kt + 1 < nk) loadTiles(kt + 1);   // latency hidden under compute

#pragma unroll
        for (int kk = 0; kk < BK; kk += 32) {
            // A fragments: 16x32 bf16, CDNA5 lane layout
            v16bf afrag[2];
#pragma unroll
            for (int mf = 0; mf < 2; ++mf) {
                const int row = waveM * 32 + mf * 16 + l16;
                const __bf16* p = &ldsA[row * LDSS + kk + half * 8];
                v8bf lo = *(const v8bf*)p;          // K: half*8 .. +7
                v8bf hi = *(const v8bf*)(p + 16);   // K: 16+half*8 .. +7
                afrag[mf] = __builtin_shufflevector(lo, hi,
                    0,1,2,3,4,5,6,7,8,9,10,11,12,13,14,15);
            }
            // B fragments: 32x16 bf16, same K order as A (half*8 .. +7, 16+half*8 .. +7)
            v16bf bfrag[4];
#pragma unroll
            for (int nf = 0; nf < 4; ++nf) {
                const int n = waveN * 64 + nf * 16 + l16;
                const __bf16* p = &ldsB[n * LDSS + kk + half * 8];
                v8bf lo = *(const v8bf*)p;
                v8bf hi = *(const v8bf*)(p + 16);
                bfrag[nf] = __builtin_shufflevector(lo, hi,
                    0,1,2,3,4,5,6,7,8,9,10,11,12,13,14,15);
            }
#pragma unroll
            for (int mf = 0; mf < 2; ++mf)
#pragma unroll
                for (int nf = 0; nf < 4; ++nf)
                    acc[mf][nf] = __builtin_amdgcn_wmma_f32_16x16x32_bf16(
                        false, afrag[mf], false, bfrag[nf],
                        (short)0, acc[mf][nf], false, false);
        }
    }

    // ---- epilogue: D layout lane L, vgpr r -> (M = r + 8*half, N = l16) ----
#pragma unroll
    for (int nf = 0; nf < 4; ++nf) {
        const int col = nBase + waveN * 64 + nf * 16 + l16;
        const float bv = bias[col];
#pragma unroll
        for (int mf = 0; mf < 2; ++mf) {
            const int rbase = mBase + waveM * 32 + mf * 16 + half * 8;
            float* op = out + (size_t)rbase * N_TOT + col;
#pragma unroll
            for (int r = 0; r < 8; ++r)
                op[(size_t)r * N_TOT] = acc[mf][nf][r] + bv;
        }
    }
}

extern "C" void kernel_launch(void* const* d_in, const int* in_sizes, int n_in,
                              void* d_out, int out_size, void* d_ws, size_t ws_size,
                              hipStream_t stream) {
    const float* x    = (const float*)d_in[0];
    const int*   wq   = (const int*)  d_in[1];
    const float* sc   = (const float*)d_in[2];
    const float* zp   = (const float*)d_in[3];
    const float* bias = (const float*)d_in[4];
    float* out = (float*)d_out;

    dim3 grid(N_TOT / BN, M_TOT / BM);   // 32 x 64 blocks
    qlin_wmma_bf16_kernel<<<grid, dim3(256), 0, stream>>>(x, wq, sc, zp, bias, out);
}
